// LinearAttention_32263794327901
// MI455X (gfx1250) — matrix operand order repaired and checked by reading, not compiled, not run on hardware
//
#include <hip/hip_runtime.h>
#include <stdint.h>

typedef __attribute__((ext_vector_type(16))) __bf16 v16bf;
typedef __attribute__((ext_vector_type(8)))  float  v8f;
typedef int v4i __attribute__((vector_size(16)));   // matches builtin param type

union FragU { v16bf f; uint4 u[2]; };

// ---- CDNA5 async global->LDS (probe-confirmed present; param0 = global int4*)
#if defined(__has_builtin)
#  if __has_builtin(__builtin_amdgcn_global_load_async_to_lds_b128)
#    define HAS_ASYNC_LDS 1
#  endif
#endif
#ifndef HAS_ASYNC_LDS
#  define HAS_ASYNC_LDS 0
#endif

#if HAS_ASYNC_LDS
#  if __has_builtin(__builtin_amdgcn_s_wait_asynccnt)
#    define WAIT_ASYNCCNT(n) __builtin_amdgcn_s_wait_asynccnt(n)
#  else
#    define WAIT_ASYNCCNT(n) asm volatile("s_wait_asynccnt %0" ::"n"(n) : "memory")
#  endif
__device__ __forceinline__ void async_copy_b128(void* lds, const void* gmem) {
  __builtin_amdgcn_global_load_async_to_lds_b128(
      (__attribute__((address_space(1))) v4i*)(gmem),
      (__attribute__((address_space(3))) v4i*)(lds), 0, 0);
}
#endif

__device__ __forceinline__ uint16_t f2bf(float x) {
  uint32_t u = __float_as_uint(x);
  u += 0x7fffu + ((u >> 16) & 1u);          // round-to-nearest-even bf16
  return (uint16_t)(u >> 16);
}
__device__ __forceinline__ float bf2f(uint16_t h) {
  return __uint_as_float(((uint32_t)h) << 16);
}
__device__ __forceinline__ v8f v8f_zero() {
  v8f z;
#pragma unroll
  for (int i = 0; i < 8; i++) z[i] = 0.f;
  return z;
}

// ---------------------------------------------------------------- fp32 -> bf16
__global__ void cvt_f32_bf16(const float* __restrict__ src,
                             uint16_t* __restrict__ dst, int n) {
  int i = (blockIdx.x * 256 + threadIdx.x) * 4;
  if (i >= n) return;
  float4 f = *(const float4*)(src + i);
  ushort4 o;
  o.x = f2bf(f.x); o.y = f2bf(f.y); o.z = f2bf(f.z); o.w = f2bf(f.w);
  *(ushort4*)(dst + i) = o;
}

__global__ void zero_f32(float* __restrict__ p, int n) {
  int i = blockIdx.x * 256 + threadIdx.x;
  if (i < n) p[i] = 0.f;
}

// ---------------------------------------------------------------- big GEMM
// Y[m,n] = sum_k A[m,k] * Wm[n,k]    (A: [16384 x 1024] bf16, Wm: [1024 x 1024] bf16)
// mode 0: store bf16 | mode 1: store bf16, elu(x)+1 | mode 2: store fp32, + bias
#define LDSTRIDE 40   // 128x32 rows padded to 40 elems (80B) to dodge bank conflicts

__device__ __forceinline__ void wmma_tile_step(
    const uint16_t* __restrict__ xs, const uint16_t* __restrict__ wsd,
    int wm, int wn, int lm, int lh, v8f (&acc)[2][4]) {
  FragU a[2], b[4];
#pragma unroll
  for (int mi = 0; mi < 2; mi++) {   // A 16x32 bf16: half-lanes K 0-7/16-23 vs 8-15/24-31
    int base = (wm * 32 + mi * 16 + lm) * LDSTRIDE + lh * 8;
    a[mi].u[0] = *(const uint4*)&xs[base];
    a[mi].u[1] = *(const uint4*)&xs[base + 16];
  }
#pragma unroll
  for (int ni = 0; ni < 4; ni++) {   // B 32x16 bf16: half-lanes K 0-7/16-23 vs 8-15/24-31
    int base = (wn * 64 + ni * 16 + lm) * LDSTRIDE + lh * 8;
    b[ni].u[0] = *(const uint4*)&wsd[base];
    b[ni].u[1] = *(const uint4*)&wsd[base + 16];
  }
#pragma unroll
  for (int mi = 0; mi < 2; mi++)
#pragma unroll
    for (int ni = 0; ni < 4; ni++)
      acc[mi][ni] = __builtin_amdgcn_wmma_f32_16x16x32_bf16(
          false, a[mi].f, false, b[ni].f, (short)0, acc[mi][ni], false, false);
}

__global__ __launch_bounds__(256) void gemm_bf16_wmma(
    const uint16_t* __restrict__ A, const uint16_t* __restrict__ Wm,
    uint16_t* __restrict__ Obf, float* __restrict__ Of32,
    const float* __restrict__ bias, int mode) {
#if HAS_ASYNC_LDS
  __shared__ uint16_t Xs[2][128 * LDSTRIDE];   // double-buffered (async pipe)
  __shared__ uint16_t Ws[2][128 * LDSTRIDE];
#else
  __shared__ uint16_t Xs[1][128 * LDSTRIDE];   // single buffer, register-staged pipe
  __shared__ uint16_t Ws[1][128 * LDSTRIDE];
#endif

  const int tid  = threadIdx.x;
  const int lane = tid & 31;
  const int wave = tid >> 5;           // 8 waves (wave32)
  const int lm = lane & 15;
  const int lh = lane >> 4;
  const int wm = wave >> 1;            // 0..3 : 32-row strip
  const int wn = wave & 1;             // 0..1 : 64-col strip

  const int m0 = blockIdx.y * 128;
  const int n0 = blockIdx.x * 128;

  v8f acc[2][4];
#pragma unroll
  for (int mi = 0; mi < 2; mi++)
#pragma unroll
    for (int ni = 0; ni < 4; ni++) acc[mi][ni] = v8f_zero();

  // staging: 128x32 bf16 tile = 512 chunks of 8 elems; each thread moves 2 per matrix
  const int rs = tid >> 2;             // 0..63
  const int cc = (tid & 3) * 8;
  const uint16_t* gA = A  + (size_t)(m0 + rs) * 1024 + cc;
  const uint16_t* gW = Wm + (size_t)(n0 + rs) * 1024 + cc;
  const int l0 = rs * LDSTRIDE + cc;
  const int l1 = (rs + 64) * LDSTRIDE + cc;

#if HAS_ASYNC_LDS
  // prologue: tile 0 -> buffer 0 (ASYNCcnt-tracked, no VGPR round trip)
  async_copy_b128(&Xs[0][l0], gA);
  async_copy_b128(&Xs[0][l1], gA + (size_t)64 * 1024);
  async_copy_b128(&Ws[0][l0], gW);
  async_copy_b128(&Ws[0][l1], gW + (size_t)64 * 1024);
  for (int it = 0; it < 32; ++it) {
    const int cur = it & 1;
    __syncthreads();                   // all waves done reading buffer cur^1
    if (it < 31) {
      const uint16_t* a2 = gA + (it + 1) * 32;
      const uint16_t* w2 = gW + (it + 1) * 32;
      async_copy_b128(&Xs[cur ^ 1][l0], a2);
      async_copy_b128(&Xs[cur ^ 1][l1], a2 + (size_t)64 * 1024);
      async_copy_b128(&Ws[cur ^ 1][l0], w2);
      async_copy_b128(&Ws[cur ^ 1][l1], w2 + (size_t)64 * 1024);
      WAIT_ASYNCCNT(4);                // 4 new in flight; current buffer landed
    } else {
      WAIT_ASYNCCNT(0);
    }
    __syncthreads();                   // current buffer visible to all waves
    wmma_tile_step(Xs[cur], Ws[cur], wm, wn, lm, lh, acc);
  }
#else
  // register-staged pipeline: issue next tile's global loads during compute
  uint4 sx0 = *(const uint4*)gA;
  uint4 sx1 = *(const uint4*)(gA + (size_t)64 * 1024);
  uint4 sw0 = *(const uint4*)gW;
  uint4 sw1 = *(const uint4*)(gW + (size_t)64 * 1024);
  for (int it = 0; it < 32; ++it) {
    __syncthreads();                   // previous compute done reading LDS
    *(uint4*)&Xs[0][l0] = sx0;  *(uint4*)&Xs[0][l1] = sx1;
    *(uint4*)&Ws[0][l0] = sw0;  *(uint4*)&Ws[0][l1] = sw1;
    __syncthreads();
    if (it < 31) {
      const int k0 = (it + 1) * 32;
      sx0 = *(const uint4*)(gA + k0);
      sx1 = *(const uint4*)(gA + k0 + (size_t)64 * 1024);
      sw0 = *(const uint4*)(gW + k0);
      sw1 = *(const uint4*)(gW + k0 + (size_t)64 * 1024);
      __builtin_prefetch(gA + k0 + 32, 0, 1);   // global_prefetch_b8
      __builtin_prefetch(gW + k0 + 32, 0, 1);
    }
    wmma_tile_step(Xs[0], Ws[0], wm, wn, lm, lh, acc);
  }
#endif

#pragma unroll
  for (int mi = 0; mi < 2; mi++) {
#pragma unroll
    for (int ni = 0; ni < 4; ni++) {
      int gn = n0 + wn * 64 + ni * 16 + lm;
#pragma unroll
      for (int j = 0; j < 8; j++) {    // C layout: VGPR j -> M=j (+8 for upper half)
        int gm = m0 + wm * 32 + mi * 16 + lh * 8 + j;
        float val = acc[mi][ni][j];
        if (mode == 1) val = (val > 0.f) ? (val + 1.f) : __expf(val);  // elu+1
        if (mode == 2) {
          Of32[(size_t)gm * 1024 + gn] = val + bias[gn];
        } else {
          Obf[(size_t)gm * 1024 + gn] = f2bf(val);
        }
      }
    }
  }
}

// ---------------------------------------------------------------- kv + k_sum
// kvT[bh][e][d] += sum_s kf[s,d]*v[s,e];  ksum[bh][d] += sum_s kf[s,d]
__global__ __launch_bounds__(256) void kv_ksum_kernel(
    const uint16_t* __restrict__ kf, const uint16_t* __restrict__ vv,
    float* __restrict__ kvT, float* __restrict__ ksum) {
  __shared__ float kfs[8][64];
  __shared__ float vs[8][64];
  const int tid = threadIdx.x;
  const int bh = blockIdx.y;
  const int b = bh >> 4, h = bh & 15;
  const size_t rowbase = (size_t)b * 4096 + (size_t)blockIdx.x * 512;
  const int colbase = h * 64;

  const int d0 = (tid >> 4) * 4;
  const int e0 = (tid & 15) * 4;
  float acc[4][4] = {};
  float ks[4] = {};
  const bool doks = (e0 == 0);

  for (int s8 = 0; s8 < 64; s8++) {
    __syncthreads();
#pragma unroll
    for (int i = 0; i < 2; i++) {
      int idx = tid + i * 256;
      int r = idx >> 6, c = idx & 63;
      size_t g = (rowbase + s8 * 8 + r) * 1024 + colbase + c;
      kfs[r][c] = bf2f(kf[g]);
      vs[r][c]  = bf2f(vv[g]);
    }
    __syncthreads();
#pragma unroll
    for (int ss = 0; ss < 8; ss++) {
      float kd[4], ve[4];
#pragma unroll
      for (int i = 0; i < 4; i++) kd[i] = kfs[ss][d0 + i];
#pragma unroll
      for (int j = 0; j < 4; j++) ve[j] = vs[ss][e0 + j];
#pragma unroll
      for (int i = 0; i < 4; i++)
#pragma unroll
        for (int j = 0; j < 4; j++) acc[i][j] += kd[i] * ve[j];
      if (doks)
#pragma unroll
        for (int i = 0; i < 4; i++) ks[i] += kd[i];
    }
  }
  float* kvbase = kvT + (size_t)bh * 4096;
#pragma unroll
  for (int i = 0; i < 4; i++)
#pragma unroll
    for (int j = 0; j < 4; j++)
      atomicAdd(&kvbase[(e0 + j) * 64 + (d0 + i)], acc[i][j]);
  if (doks)
#pragma unroll
    for (int i = 0; i < 4; i++) atomicAdd(&ksum[bh * 64 + d0 + i], ks[i]);
}

// ---------------------------------------------------------------- qf @ kv / denom
#define KVSTRIDE 72
__global__ __launch_bounds__(256) void qkv_ctx_kernel(
    const uint16_t* __restrict__ qf, const float* __restrict__ kvT,
    const float* __restrict__ ksum, uint16_t* __restrict__ ctx) {
  __shared__ uint16_t kvs[64 * KVSTRIDE];
  __shared__ float ksum_s[64];
  __shared__ float denom_s[8][16];

  const int tid = threadIdx.x;
  const int lane = tid & 31;
  const int wave = tid >> 5;
  const int lm = lane & 15, lh = lane >> 4;
  const int bh = blockIdx.y;
  const int b = bh >> 4, h = bh & 15;
  const size_t grow0 = (size_t)b * 4096 + (size_t)blockIdx.x * 128;

  // kvT[e,d] fp32 -> LDS bf16; rows of kvs are the B^T rows (lane n contiguous)
  for (int i = tid; i < 4096; i += 256) {
    int e = i >> 6, d = i & 63;
    kvs[e * KVSTRIDE + d] = f2bf(kvT[(size_t)bh * 4096 + i]);
  }
  if (tid < 64) ksum_s[tid] = ksum[bh * 64 + tid];
  __syncthreads();

  const size_t grow = grow0 + wave * 16 + lm;
  const uint16_t* qrow = qf + grow * 1024 + h * 64;

  // denom[row] = dot(qf[row,:], k_sum) + eps ; each half-lane does 32 of 64 terms
  float part = 0.f;
#pragma unroll
  for (int c = 0; c < 4; c++) {
    uint4 u = *(const uint4*)(qrow + lh * 32 + c * 8);
    uint32_t w[4] = {u.x, u.y, u.z, u.w};
#pragma unroll
    for (int t = 0; t < 4; t++) {
      part += bf2f((uint16_t)(w[t] & 0xffffu)) * ksum_s[lh * 32 + c * 8 + t * 2];
      part += bf2f((uint16_t)(w[t] >> 16))     * ksum_s[lh * 32 + c * 8 + t * 2 + 1];
    }
  }
  part += __shfl_xor(part, 16, 32);
  if (lh == 0) denom_s[wave][lm] = part + 1e-6f;
  __syncthreads();

  v8f acc[4];
#pragma unroll
  for (int ni = 0; ni < 4; ni++) acc[ni] = v8f_zero();
#pragma unroll
  for (int k0 = 0; k0 < 64; k0 += 32) {
    FragU a;
    a.u[0] = *(const uint4*)(qrow + k0 + lh * 8);
    a.u[1] = *(const uint4*)(qrow + k0 + 16 + lh * 8);
#pragma unroll
    for (int ni = 0; ni < 4; ni++) {
      FragU bfr;
      int base = (ni * 16 + lm) * KVSTRIDE + k0 + lh * 8;
      bfr.u[0] = *(const uint4*)&kvs[base];
      bfr.u[1] = *(const uint4*)&kvs[base + 16];
      acc[ni] = __builtin_amdgcn_wmma_f32_16x16x32_bf16(
          false, a.f, false, bfr.f, (short)0, acc[ni], false, false);
    }
  }

#pragma unroll
  for (int ni = 0; ni < 4; ni++) {
    int gn = h * 64 + ni * 16 + lm;
#pragma unroll
    for (int j = 0; j < 8; j++) {
      int m = lh * 8 + j;
      size_t gm = grow0 + wave * 16 + m;
      float val = acc[ni][j] / denom_s[wave][m];
      ctx[gm * 1024 + gn] = f2bf(val);
    }
  }
}

// ---------------------------------------------------------------- launcher
extern "C" void kernel_launch(void* const* d_in, const int* in_sizes, int n_in,
                              void* d_out, int out_size, void* d_ws, size_t ws_size,
                              hipStream_t stream) {
  (void)in_sizes; (void)n_in; (void)out_size; (void)ws_size;
  const float* x  = (const float*)d_in[0];
  const float* wq = (const float*)d_in[1];
  const float* wk = (const float*)d_in[2];
  const float* wv = (const float*)d_in[3];
  const float* wo = (const float*)d_in[4];
  const float* bo = (const float*)d_in[5];

  const size_t M = 16384, D = 1024;
  char* ws = (char*)d_ws;
  size_t off = 0;
  uint16_t* xb  = (uint16_t*)(ws + off); off += M * D * 2;   // reused as ctx later
  uint16_t* wqb = (uint16_t*)(ws + off); off += D * D * 2;
  uint16_t* wkb = (uint16_t*)(ws + off); off += D * D * 2;
  uint16_t* wvb = (uint16_t*)(ws + off); off += D * D * 2;
  uint16_t* wob = (uint16_t*)(ws + off); off += D * D * 2;
  uint16_t* qfb = (uint16_t*)(ws + off); off += M * D * 2;
  uint16_t* kfb = (uint16_t*)(ws + off); off += M * D * 2;
  uint16_t* vvb = (uint16_t*)(ws + off); off += M * D * 2;
  float*    kvT = (float*)(ws + off);    off += (size_t)64 * 4096 * 4;
  float*    ksm = (float*)(ws + off);    off += (size_t)64 * 64 * 4;   // contiguous after kvT
  uint16_t* ctx = xb;

  cvt_f32_bf16<<<(int)(M * D / 1024), 256, 0, stream>>>(x,  xb,  (int)(M * D));
  cvt_f32_bf16<<<(int)(D * D / 1024), 256, 0, stream>>>(wq, wqb, (int)(D * D));
  cvt_f32_bf16<<<(int)(D * D / 1024), 256, 0, stream>>>(wk, wkb, (int)(D * D));
  cvt_f32_bf16<<<(int)(D * D / 1024), 256, 0, stream>>>(wv, wvb, (int)(D * D));
  cvt_f32_bf16<<<(int)(D * D / 1024), 256, 0, stream>>>(wo, wob, (int)(D * D));
  zero_f32<<<(64 * 4096 + 64 * 64 + 255) / 256, 256, 0, stream>>>(kvT, 64 * 4096 + 64 * 64);

  dim3 gg(D / 128, M / 128);  // (8, 128)
  gemm_bf16_wmma<<<gg, 256, 0, stream>>>(xb, wqb, qfb, nullptr, nullptr, 1);
  gemm_bf16_wmma<<<gg, 256, 0, stream>>>(xb, wkb, kfb, nullptr, nullptr, 1);
  gemm_bf16_wmma<<<gg, 256, 0, stream>>>(xb, wvb, vvb, nullptr, nullptr, 0);

  kv_ksum_kernel<<<dim3(8, 64), 256, 0, stream>>>(kfb, vvb, kvT, ksm);
  qkv_ctx_kernel<<<dim3(32, 64), 256, 0, stream>>>(qfb, kvT, ksm, ctx);

  gemm_bf16_wmma<<<gg, 256, 0, stream>>>(ctx, wob, nullptr, (float*)d_out, bo, 2);
}
